// GeomGCN_30640296689801
// MI455X (gfx1250) — matrix operand order repaired and checked by reading, not compiled, not run on hardware
//
#include <hip/hip_runtime.h>
#include <hip/hip_bf16.h>
#include <math.h>

// CDNA5 / gfx1250, wave32. Matrix math via v_wmma_f32_16x16x32_bf16.
typedef __attribute__((ext_vector_type(16))) __bf16 v16bf;
typedef __attribute__((ext_vector_type(8)))  float  v8f;

#define NUM_REL 4

// ---------------------------------------------------------------- degree
__global__ void deg_kernel(const int* __restrict__ row, float* __restrict__ deg, int E) {
    int e = blockIdx.x * blockDim.x + threadIdx.x;
    if (e < E) atomicAdd(&deg[row[e]], 1.0f);
}

// deg -> deg^-1/2 in place
__global__ void dinv_kernel(float* __restrict__ deg, int N) {
    int n = blockIdx.x * blockDim.x + threadIdx.x;
    if (n < N) {
        float d = deg[n];
        deg[n] = (d > 0.0f) ? rsqrtf(d) : 0.0f;
    }
}

__global__ void norm_kernel(const int* __restrict__ row, const int* __restrict__ col,
                            const float* __restrict__ dinv, float* __restrict__ norm, int E) {
    int e = blockIdx.x * blockDim.x + threadIdx.x;
    if (e < E) norm[e] = dinv[row[e]] * dinv[col[e]];
}

// ------------------------------------------------- message scatter (segment_sum)
// agg[(rel*N + row)*F + f] += src[col*F + f] * norm.  One thread = 4 features of one edge.
__global__ void scatter_kernel(const float* __restrict__ src,
                               const int* __restrict__ row, const int* __restrict__ col,
                               const int* __restrict__ rel, const float* __restrict__ norm,
                               float* __restrict__ agg,
                               int E, int Fdim, int vecShift, int Nnodes) {
    int t = blockIdx.x * blockDim.x + threadIdx.x;
    int e = t >> vecShift;
    if (e >= E) return;
    int j = (t & ((1 << vecShift) - 1)) << 2;
    float nrm = norm[e];
    const float4 v = *reinterpret_cast<const float4*>(src + (size_t)col[e] * Fdim + j);
    float* dst = agg + ((size_t)rel[e] * Nnodes + row[e]) * (size_t)Fdim + j;
    atomicAdd(dst + 0, v.x * nrm);
    atomicAdd(dst + 1, v.y * nrm);
    atomicAdd(dst + 2, v.z * nrm);
    atomicAdd(dst + 3, v.w * nrm);
}

// --------------------------------------- pack W [K, Ncols] f32 -> bf16 B fragments
// B-matrix 32x16 (16-bit) per-lane layout (ISA 7.12.2): lane = (half<<4)|n,
// element j of lane holds B[kt*32 + half*8 + j + (j>=8 ? 8 : 0)][nt*16 + n].
// Stored contiguously: Wp[(((kt*NT)+nt)*32 + lane)*16 + j]
__global__ void pack_w_bf16(const float* __restrict__ W, __bf16* __restrict__ Wp,
                            int K, int Ncols) {
    int t = blockIdx.x * blockDim.x + threadIdx.x;
    int NT = Ncols >> 4;
    int KT = K >> 5;
    int total = KT * NT * 32;
    if (t >= total) return;
    int lane = t & 31;
    int nt   = (t >> 5) % NT;
    int kt   = (t >> 5) / NT;
    int n     = nt * 16 + (lane & 15);
    int kbase = kt * 32 + ((lane >> 4) << 3);
    __bf16* dst = Wp + (size_t)t * 16;
#pragma unroll
    for (int j = 0; j < 16; ++j)
        dst[j] = (__bf16)W[(size_t)(kbase + j + ((j >> 3) << 3)) * Ncols + n];
}

// ------------------------------------------------------- WMMA GEMM over rel panels
// out[N, Ncols] = concat_rel(agg)[N, K] @ W + bias, K = NUM_REL*Fdim.
// One wave -> one 16x16 output tile. A logical element (n, k) lives at
// agg[((k>>Fshift)*N + n)*Fdim + (k & (Fdim-1))]; each 32-wide K tile stays in one panel.
__global__ void __launch_bounds__(32)
rgcn_wmma_gemm(const float* __restrict__ agg, int Nnodes, int Fdim, int Fshift,
               const __bf16* __restrict__ Wpack, const float* __restrict__ bias,
               float* __restrict__ out, int Ncols, int K) {
    const int lane = threadIdx.x & 31;
    const int m    = lane & 15;
    const int hk   = (lane >> 4) << 3;          // 0 or 8 (A-matrix half select)
    const int tileRow = blockIdx.x;
    const int nt      = blockIdx.y;
    const int NT      = Ncols >> 4;
    int row = tileRow * 16 + m;
    if (row >= Nnodes) row = Nnodes - 1;        // clamp for (unused) partial tiles

    v8f acc = {};
    const int KT = K >> 5;
#pragma unroll 2
    for (int kt = 0; kt < KT; ++kt) {
        const int kbase = kt << 5;
        const int panel = kbase >> Fshift;
        const int fr    = kbase & (Fdim - 1);
        const float* Abase = agg + ((size_t)panel * Nnodes + row) * (size_t)Fdim + fr;
        // A fragment: lane holds row m, K = hk..hk+7 and 16+hk..16+hk+7
        const float4 a0 = *reinterpret_cast<const float4*>(Abase + hk);
        const float4 a1 = *reinterpret_cast<const float4*>(Abase + hk + 4);
        const float4 a2 = *reinterpret_cast<const float4*>(Abase + 16 + hk);
        const float4 a3 = *reinterpret_cast<const float4*>(Abase + 20 + hk);
        v16bf a;
        a[0]  = (__bf16)a0.x; a[1]  = (__bf16)a0.y; a[2]  = (__bf16)a0.z; a[3]  = (__bf16)a0.w;
        a[4]  = (__bf16)a1.x; a[5]  = (__bf16)a1.y; a[6]  = (__bf16)a1.z; a[7]  = (__bf16)a1.w;
        a[8]  = (__bf16)a2.x; a[9]  = (__bf16)a2.y; a[10] = (__bf16)a2.z; a[11] = (__bf16)a2.w;
        a[12] = (__bf16)a3.x; a[13] = (__bf16)a3.y; a[14] = (__bf16)a3.z; a[15] = (__bf16)a3.w;
        // B fragment: prepacked, one 32B contiguous load per lane
        const v16bf b = *reinterpret_cast<const v16bf*>(
            Wpack + (((size_t)kt * NT + nt) * 32 + lane) * 16);
        acc = __builtin_amdgcn_wmma_f32_16x16x32_bf16(
            false, a, false, b, (short)0, acc, false, false);
    }
    // C/D layout: VGPR r -> M = (lane>>4)*8 + r, N = lane&15
    const int outCol = nt * 16 + (lane & 15);
    const int rbase  = tileRow * 16 + ((lane >> 4) << 3);
    const float bv = bias[outCol];
    float* op = out + (size_t)rbase * Ncols + outCol;
    if (rbase + 8 <= Nnodes) {
        // full tile (always the case for N % 16 == 0): straight-line stores
#pragma unroll
        for (int r = 0; r < 8; ++r) { *op = acc[r] + bv; op += Ncols; }
    } else {
#pragma unroll
        for (int r = 0; r < 8; ++r) {
            if (rbase + r < Nnodes) *op = acc[r] + bv;
            op += Ncols;
        }
    }
}

// -------------------------------------------------------------- log_softmax (C==16 fast path)
__global__ void logsoftmax_kernel(float* __restrict__ out, int N, int C) {
    int n = blockIdx.x * blockDim.x + threadIdx.x;
    if (n >= N) return;
    float* r = out + (size_t)n * C;
    if (C == 16) {
        float4 v0 = *reinterpret_cast<const float4*>(r + 0);
        float4 v1 = *reinterpret_cast<const float4*>(r + 4);
        float4 v2 = *reinterpret_cast<const float4*>(r + 8);
        float4 v3 = *reinterpret_cast<const float4*>(r + 12);
        float mx = fmaxf(fmaxf(fmaxf(v0.x, v0.y), fmaxf(v0.z, v0.w)),
                   fmaxf(fmaxf(fmaxf(v1.x, v1.y), fmaxf(v1.z, v1.w)),
                   fmaxf(fmaxf(fmaxf(v2.x, v2.y), fmaxf(v2.z, v2.w)),
                         fmaxf(fmaxf(v3.x, v3.y), fmaxf(v3.z, v3.w)))));
        float s = expf(v0.x - mx) + expf(v0.y - mx) + expf(v0.z - mx) + expf(v0.w - mx)
                + expf(v1.x - mx) + expf(v1.y - mx) + expf(v1.z - mx) + expf(v1.w - mx)
                + expf(v2.x - mx) + expf(v2.y - mx) + expf(v2.z - mx) + expf(v2.w - mx)
                + expf(v3.x - mx) + expf(v3.y - mx) + expf(v3.z - mx) + expf(v3.w - mx);
        float sh = mx + logf(s);
        v0.x -= sh; v0.y -= sh; v0.z -= sh; v0.w -= sh;
        v1.x -= sh; v1.y -= sh; v1.z -= sh; v1.w -= sh;
        v2.x -= sh; v2.y -= sh; v2.z -= sh; v2.w -= sh;
        v3.x -= sh; v3.y -= sh; v3.z -= sh; v3.w -= sh;
        *reinterpret_cast<float4*>(r + 0)  = v0;
        *reinterpret_cast<float4*>(r + 4)  = v1;
        *reinterpret_cast<float4*>(r + 8)  = v2;
        *reinterpret_cast<float4*>(r + 12) = v3;
    } else {
        float v[32];
        float mx = -3.402823466e38f;
        for (int c = 0; c < C && c < 32; ++c) { v[c] = r[c]; mx = fmaxf(mx, v[c]); }
        float s = 0.0f;
        for (int c = 0; c < C && c < 32; ++c) s += expf(v[c] - mx);
        float l = logf(s);
        for (int c = 0; c < C && c < 32; ++c) r[c] = v[c] - mx - l;
    }
}

// ------------------------------------------------------------------------ driver
extern "C" void kernel_launch(void* const* d_in, const int* in_sizes, int n_in,
                              void* d_out, int out_size, void* d_ws, size_t ws_size,
                              hipStream_t stream) {
    const float* x    = (const float*)d_in[0];
    const int*   eidx = (const int*)  d_in[1];   // [2,E]: rows then cols
    const int*   erel = (const int*)  d_in[2];
    const float* W1   = (const float*)d_in[3];
    const float* b1   = (const float*)d_in[4];
    const float* W2   = (const float*)d_in[5];
    const float* b2   = (const float*)d_in[6];

    const int E = in_sizes[2];
    const int H = in_sizes[4];
    const int C = in_sizes[6];
    const int F = in_sizes[3] / (NUM_REL * H);       // W1 is [4F, H]
    const int N = in_sizes[0] / F;
    const int Fshift = 31 - __builtin_clz((unsigned)F);
    const int Hshift = 31 - __builtin_clz((unsigned)H);
    const int* erow = eidx;
    const int* ecol = eidx + E;

    const int K1 = NUM_REL * F;                      // 1024
    const int K2 = NUM_REL * H;                      // 512
    const int KT1 = K1 >> 5, NT1 = H >> 4;
    const int KT2 = K2 >> 5, NT2 = C >> 4;

    // ---- workspace carve-out (all 256B aligned)
    char* base = (char*)d_ws;
    size_t off = 0;
    auto carve = [&](size_t bytes) -> char* {
        char* p = base + off;
        off = (off + bytes + 255) & ~(size_t)255;
        return p;
    };
    float*  dinv   = (float*) carve((size_t)N * 4);
    float*  norm   = (float*) carve((size_t)E * 4);
    float*  h1     = (float*) carve((size_t)N * H * 4);
    __bf16* wpack1 = (__bf16*)carve((size_t)KT1 * NT1 * 32 * 16 * 2);
    __bf16* wpack2 = (__bf16*)carve((size_t)KT2 * NT2 * 32 * 16 * 2);
    float*  agg    = (float*) carve((size_t)NUM_REL * N * F * 4);  // reused for layer 2
    (void)ws_size;

    float* out = (float*)d_out;
    const int TB = 256;

    // ---- normalization
    hipMemsetAsync(dinv, 0, (size_t)N * 4, stream);
    deg_kernel <<<(E + TB - 1) / TB, TB, 0, stream>>>(erow, dinv, E);
    dinv_kernel<<<(N + TB - 1) / TB, TB, 0, stream>>>(dinv, N);
    norm_kernel<<<(E + TB - 1) / TB, TB, 0, stream>>>(erow, ecol, dinv, norm, E);

    // ---- pack weights to bf16 B fragments
    {
        int tot1 = KT1 * NT1 * 32;
        int tot2 = KT2 * NT2 * 32;
        pack_w_bf16<<<(tot1 + TB - 1) / TB, TB, 0, stream>>>(W1, wpack1, K1, H);
        pack_w_bf16<<<(tot2 + TB - 1) / TB, TB, 0, stream>>>(W2, wpack2, K2, C);
    }

    // ---- layer 1: scatter + GEMM -> h1 [N, H]
    hipMemsetAsync(agg, 0, (size_t)NUM_REL * N * F * 4, stream);
    {
        int vecShift = Fshift - 2;                       // F/4 float4s per edge
        long long threads = (long long)E << vecShift;
        scatter_kernel<<<(unsigned)((threads + TB - 1) / TB), TB, 0, stream>>>(
            x, erow, ecol, erel, norm, agg, E, F, vecShift, N);
        dim3 grid((N + 15) / 16, NT1);
        rgcn_wmma_gemm<<<grid, 32, 0, stream>>>(agg, N, F, Fshift, wpack1, b1, h1, H, K1);
    }

    // ---- layer 2: scatter + GEMM -> out [N, C]
    hipMemsetAsync(agg, 0, (size_t)NUM_REL * N * H * 4, stream);
    {
        int vecShift = Hshift - 2;
        long long threads = (long long)E << vecShift;
        scatter_kernel<<<(unsigned)((threads + TB - 1) / TB), TB, 0, stream>>>(
            h1, erow, ecol, erel, norm, agg, E, H, vecShift, N);
        dim3 grid((N + 15) / 16, NT2);
        rgcn_wmma_gemm<<<grid, 32, 0, stream>>>(agg, N, H, Hshift, wpack2, b2, out, C, K2);
    }

    // ---- log_softmax in place on d_out
    logsoftmax_kernel<<<(N + TB - 1) / TB, TB, 0, stream>>>(out, N, C);
}
